// GL_TAGConv_3l_1024h_w_k3_gnorm_meanaggr_44753559224341
// MI455X (gfx1250) — compile-verified, matrix operand order checked by reading, not run on hardware
//
#include <hip/hip_runtime.h>
#include <cstdint>

// ---------------------------------------------------------------------------
// GL_TAGConv_3l_1024h_w_k3_gnorm_meanaggr — CDNA5 (gfx1250)
//
// Dense GEMMs (~480 GFLOP, dominated by layer2's 4x [50000,1024]x[1024,1024])
// use v_wmma_f32_16x16x32_bf16 with f32 accumulation. Weights are converted
// to bf16 *transposed* once so both A and B LDS tiles are row-major and all
// WMMA fragments load as ds_load_b128. Interior tiles run a double-buffered
// pipeline: global_load_async_to_lds_b128 prefetches tile k+1 (ASYNCcnt)
// while tile k feeds the WMMAs; one barrier per K-step. Edge tiles use
// clamped 16B vector staging. Graph hops (atomic scatter) and GraphNorm are
// bandwidth-bound f32 kernels.
// ---------------------------------------------------------------------------

typedef __attribute__((ext_vector_type(16))) __bf16          v16bf;
typedef __attribute__((ext_vector_type(8)))  float           v8f;
typedef __attribute__((ext_vector_type(8)))  unsigned short  u16x8;

#define TB_BM 128
#define TB_BN 128
#define TB_BK 32
#define TB_LD 48                        // LDS row stride in halfwords (96B)
#define TILE_HW (TB_BM * TB_LD)         // halfwords per tile buffer
#define TILE_BYTES (TILE_HW * 2)        // 12288 B

__device__ __forceinline__ unsigned short f2bf(float f) {
    unsigned int u = __float_as_uint(f);
    unsigned int r = u + 0x7FFFu + ((u >> 16) & 1u);      // round-nearest-even
    if ((u & 0x7F800000u) == 0x7F800000u) r = u;          // inf / nan: truncate
    return (unsigned short)(r >> 16);
}

// WMMA 16-bit operand fragment per ISA 7.12.2: lane<16 (hi=0) holds K in
// {0..7,16..23}; lane>=16 (hi=1) holds {8..15,24..31}. Row-major LDS tile ->
// two contiguous 16B ds_load_b128.
__device__ __forceinline__ v16bf load_frag(const unsigned short* rowp, int hi) {
    u16x8 lo = *(const u16x8*)(rowp + (hi << 3));
    u16x8 hv = *(const u16x8*)(rowp + 16 + (hi << 3));
    return __builtin_bit_cast(
        v16bf, __builtin_shufflevector(lo, hv, 0, 1, 2, 3, 4, 5, 6, 7,
                                               8, 9, 10, 11, 12, 13, 14, 15));
}

// ---------------------------------------------------------------------------
// Edge-norm precompute
// ---------------------------------------------------------------------------
__global__ void k_deg_cnt(const float* __restrict__ w, const int* __restrict__ dst,
                          float* __restrict__ deg, float* __restrict__ cnt, int E) {
    int e = blockIdx.x * 256 + threadIdx.x;
    if (e >= E) return;
    int t = dst[e];
    atomicAdd(&deg[t], w[e]);
    atomicAdd(&cnt[t], 1.0f);
}

__global__ void k_inv(float* __restrict__ deg_to_dinv, float* __restrict__ cnt_to_cinv, int N) {
    int i = blockIdx.x * 256 + threadIdx.x;
    if (i >= N) return;
    float d = deg_to_dinv[i];
    deg_to_dinv[i] = (d > 0.0f) ? rsqrtf(d) : 0.0f;
    float c = cnt_to_cinv[i];
    cnt_to_cinv[i] = (c > 0.0f) ? (1.0f / c) : 0.0f;
}

__global__ void k_norm_w(float* __restrict__ nw, const float* __restrict__ dinv,
                         const float* __restrict__ w,
                         const int* __restrict__ src, const int* __restrict__ dst, int E) {
    int e = blockIdx.x * 256 + threadIdx.x;
    if (e >= E) return;
    nw[e] = dinv[src[e]] * w[e] * dinv[dst[e]];
}

// ---------------------------------------------------------------------------
// f32 -> bf16 converts (plain for activations, per-slice transposed for W)
// ---------------------------------------------------------------------------
__global__ void k_cvt_bf16(const float* __restrict__ in, unsigned short* __restrict__ out, long n) {
    long i = (long)blockIdx.x * 256 + threadIdx.x;
    if (i < n) out[i] = f2bf(in[i]);
}

// in: [S][K][N] f32  ->  out: [S][N][K] bf16
__global__ void k_cvt_bf16_T(const float* __restrict__ in, unsigned short* __restrict__ out,
                             int K, int N, long total) {
    long i = (long)blockIdx.x * 256 + threadIdx.x;
    if (i >= total) return;
    long kn = (long)K * N;
    long s = i / kn;
    long rem = i - s * kn;
    int k = (int)(rem / N);
    int n = (int)(rem - (long)k * N);
    out[s * kn + (long)n * K + k] = f2bf(in[i]);
}

// ---------------------------------------------------------------------------
// Graph hop: out[dst] += nw * h[src]   (edge-parallel, float4 + f32 atomics)
// ---------------------------------------------------------------------------
__global__ void k_scatter(const float* __restrict__ h, float* __restrict__ out,
                          const float* __restrict__ nw,
                          const int* __restrict__ src, const int* __restrict__ dst,
                          int E, int d) {
    int e = blockIdx.x;
    int f = threadIdx.x * 4;
    if (e >= E || f >= d) return;
    int s = src[e], t = dst[e];
    float wv = nw[e];
    float4 m = *(const float4*)(h + (long)s * d + f);
    float* op = out + (long)t * d + f;
    atomicAdd(op + 0, wv * m.x);
    atomicAdd(op + 1, wv * m.y);
    atomicAdd(op + 2, wv * m.z);
    atomicAdd(op + 3, wv * m.w);
}

// h *= cnt_inv[node] (in place) + emit bf16 copy (GEMM A operand)
__global__ void k_scale_convert(float* __restrict__ h, const float* __restrict__ cinv,
                                unsigned short* __restrict__ hb, int N, int d) {
    long q = (long)blockIdx.x * 256 + threadIdx.x;
    long i = q * 4;
    if (i >= (long)N * d) return;
    float c = cinv[i / d];
    float4 v = *(const float4*)(h + i);
    v.x *= c; v.y *= c; v.z *= c; v.w *= c;
    *(float4*)(h + i) = v;
    ushort4 b;
    b.x = f2bf(v.x); b.y = f2bf(v.y); b.z = f2bf(v.z); b.w = f2bf(v.w);
    *(ushort4*)(hb + i) = b;
}

// ---------------------------------------------------------------------------
// WMMA GEMM:  C[M,Nn] = (init ? bias : C) + A_bf16[M,K] @ BT_bf16[Nn,K]^T
// 128x128 block tile, BK=32, 8 waves, wave tile 64x32 (4x2 WMMAs / K-step).
// ---------------------------------------------------------------------------
__global__ __launch_bounds__(256)
void k_gemm_bf16(const unsigned short* __restrict__ A, const unsigned short* __restrict__ BT,
                 float* __restrict__ C, const float* __restrict__ bias,
                 int M, int Nn, int K, int initC) {
    __shared__ unsigned short As[2 * TILE_HW];
    __shared__ unsigned short Bs[2 * TILE_HW];

    const int tid  = threadIdx.x;
    const int lane = tid & 31;
    const int wave = tid >> 5;
    const int r    = lane & 15;        // M-row / N-col within 16x16 frag
    const int hi   = lane >> 4;        // K-half select
    const int wm   = (wave & 1) * 64;  // wave row offset in block tile
    const int wn   = (wave >> 1) * 32; // wave col offset in block tile

    const long m0 = (long)blockIdx.x * TB_BM;
    const long n0 = (long)blockIdx.y * TB_BN;
    const bool fast = (m0 + TB_BM <= M) && (n0 + TB_BN <= Nn);

    v8f acc[4][2];
    #pragma unroll
    for (int i = 0; i < 4; ++i)
        #pragma unroll
        for (int j = 0; j < 2; ++j) {
            int col = wn + j * 16 + r;
            if (initC) {
                long cc = n0 + col;
                float bv = (cc < Nn) ? bias[cc] : 0.0f;
                #pragma unroll
                for (int v = 0; v < 8; ++v) acc[i][j][v] = bv;
            } else {
                #pragma unroll
                for (int v = 0; v < 8; ++v) {
                    long row = m0 + wm + i * 16 + v + hi * 8;
                    long cc  = n0 + col;
                    acc[i][j][v] = (row < M && cc < Nn) ? C[row * Nn + cc] : 0.0f;
                }
            }
        }

    const unsigned asA = (unsigned)(uintptr_t)&As[0];
    const unsigned asB = (unsigned)(uintptr_t)&Bs[0];
    const int srow = tid >> 2;          // 0..63 (staging row per pass)
    const int sc8  = (tid & 3) * 8;     // 8-halfword chunk within 32-wide row

    // one WMMA K-step consuming tile buffer p
    auto mma_step = [&](const unsigned short* Ap, const unsigned short* Bp) {
        v16bf af[4], bfr[2];
        #pragma unroll
        for (int i = 0; i < 4; ++i)
            af[i] = load_frag(Ap + (wm + i * 16 + r) * TB_LD, hi);
        #pragma unroll
        for (int j = 0; j < 2; ++j)
            bfr[j] = load_frag(Bp + (wn + j * 16 + r) * TB_LD, hi);
        #pragma unroll
        for (int i = 0; i < 4; ++i)
            #pragma unroll
            for (int j = 0; j < 2; ++j)
                acc[i][j] = __builtin_amdgcn_wmma_f32_16x16x32_bf16(
                    false, af[i], false, bfr[j], (short)0, acc[i][j], false, false);
    };

    if (fast) {
        // async stage of one 128x32 A tile + 128x32 B tile into buffer pb
        auto stage_async = [&](int kk, int pb) {
            #pragma unroll
            for (int p = 0; p < 2; ++p) {
                int row = srow + p * 64;
                unsigned la = asA + (unsigned)pb * TILE_BYTES +
                              (unsigned)(row * TB_LD + sc8) * 2u;
                const unsigned short* ga = A + (m0 + row) * K + kk + sc8;
                asm volatile("global_load_async_to_lds_b128 %0, %1, off"
                             :: "v"(la), "v"((unsigned long long)(uintptr_t)ga)
                             : "memory");
                unsigned lb = asB + (unsigned)pb * TILE_BYTES +
                              (unsigned)(row * TB_LD + sc8) * 2u;
                const unsigned short* gb = BT + (n0 + row) * K + kk + sc8;
                asm volatile("global_load_async_to_lds_b128 %0, %1, off"
                             :: "v"(lb), "v"((unsigned long long)(uintptr_t)gb)
                             : "memory");
            }
        };

        const int nsteps = K >> 5;
        stage_async(0, 0);
        for (int s = 0; s < nsteps; ++s) {
            // my tile-s writes done; barrier => everyone's are done
            asm volatile("s_wait_asynccnt 0x0" ::: "memory");
            __syncthreads();
            if (s + 1 < nsteps) stage_async((s + 1) << 5, (s + 1) & 1);
            mma_step(&As[(s & 1) * TILE_HW], &Bs[(s & 1) * TILE_HW]);
            // all my LDS reads of buffer s&1 complete before next barrier,
            // so tile s+2's async writes (issued after it) cannot race them
            asm volatile("s_wait_dscnt 0x0" ::: "memory");
        }
    } else {
        // edge tiles: clamped-row 16B vector staging (K always full)
        for (int kk = 0; kk < K; kk += TB_BK) {
            #pragma unroll
            for (int p = 0; p < 2; ++p) {
                int row = srow + p * 64;
                u16x8 z = {};
                long gr = m0 + row;
                bool okA = gr < M;
                u16x8 va = *(const u16x8*)(A + (okA ? gr : (long)(M - 1)) * K + kk + sc8);
                *(u16x8*)&As[row * TB_LD + sc8] = okA ? va : z;
                long gc = n0 + row;
                bool okB = gc < Nn;
                u16x8 vb = *(const u16x8*)(BT + (okB ? gc : (long)(Nn - 1)) * K + kk + sc8);
                *(u16x8*)&Bs[row * TB_LD + sc8] = okB ? vb : z;
            }
            __syncthreads();
            mma_step(&As[0], &Bs[0]);
            __syncthreads();
        }
    }

    // store per ISA 16x16 f32 C/D layout: VGPR v -> row v + hi*8, col = r
    #pragma unroll
    for (int i = 0; i < 4; ++i)
        #pragma unroll
        for (int j = 0; j < 2; ++j) {
            long cc = n0 + wn + j * 16 + r;
            if (cc >= Nn) continue;
            #pragma unroll
            for (int v = 0; v < 8; ++v) {
                long row = m0 + wm + i * 16 + v + hi * 8;
                if (row < M) C[row * Nn + cc] = acc[i][j][v];
            }
        }
}

// ---------------------------------------------------------------------------
// ELU + GraphNorm
// ---------------------------------------------------------------------------
__global__ void k_elu(float* __restrict__ h, long n) {
    long i = (long)blockIdx.x * 256 + threadIdx.x;
    if (i >= n) return;
    float x = h[i];
    h[i] = (x > 0.0f) ? x : expm1f(x);
}

#define GN_ROWS 256
__global__ void k_colsum(const float* __restrict__ h, float* __restrict__ sum, int N, int d) {
    int col = blockIdx.x * 256 + threadIdx.x;
    int r0 = blockIdx.y * GN_ROWS;
    int r1 = min(r0 + GN_ROWS, N);
    float local = 0.0f;
    for (int r = r0; r < r1; ++r) local += h[(long)r * d + col];
    atomicAdd(&sum[col], local);
}

__global__ void k_colvar(const float* __restrict__ h, const float* __restrict__ sum,
                         const float* __restrict__ a, float* __restrict__ var, int N, int d) {
    int col = blockIdx.x * 256 + threadIdx.x;
    float am = a[col] * (sum[col] / (float)N);
    int r0 = blockIdx.y * GN_ROWS;
    int r1 = min(r0 + GN_ROWS, N);
    float local = 0.0f;
    for (int r = r0; r < r1; ++r) {
        float o = h[(long)r * d + col] - am;
        local += o * o;
    }
    atomicAdd(&var[col], local);
}

__global__ void k_gnorm_apply(const float* __restrict__ h, const float* __restrict__ sum,
                              const float* __restrict__ var, const float* __restrict__ g,
                              const float* __restrict__ be, const float* __restrict__ a,
                              float* __restrict__ outF, unsigned short* __restrict__ outB,
                              int N, int d) {
    long i = (long)blockIdx.x * 256 + threadIdx.x;
    if (i >= (long)N * d) return;
    int col = (int)(i % d);
    float mean = sum[col] / (float)N;
    float v    = var[col] / (float)N;
    float o = h[i] - a[col] * mean;
    float y = g[col] * o * rsqrtf(v + 1e-5f) + be[col];
    outF[i] = y;
    outB[i] = f2bf(y);
}

// ---------------------------------------------------------------------------
// Host orchestration
// ---------------------------------------------------------------------------
extern "C" void kernel_launch(void* const* d_in, const int* in_sizes, int n_in,
                              void* d_out, int out_size, void* d_ws, size_t ws_size,
                              hipStream_t stream) {
    (void)n_in; (void)out_size; (void)ws_size;

    const int FIN = 128, H = 1024, Cc = 16;
    const int N = in_sizes[0] / FIN;       // 50000
    const int E = in_sizes[2];             // 200000

    const float* x   = (const float*)d_in[0];
    const int*   ei  = (const int*)d_in[1];          // [2,E] int32
    const float* w   = (const float*)d_in[2];
    const float* W1  = (const float*)d_in[3];
    const float* b1  = (const float*)d_in[4];
    const float* g1  = (const float*)d_in[5];
    const float* be1 = (const float*)d_in[6];
    const float* a1  = (const float*)d_in[7];
    const float* W2  = (const float*)d_in[8];
    const float* b2  = (const float*)d_in[9];
    const float* g2  = (const float*)d_in[10];
    const float* be2 = (const float*)d_in[11];
    const float* a2  = (const float*)d_in[12];
    const float* W3  = (const float*)d_in[13];
    const float* b3  = (const float*)d_in[14];
    const int* src = ei;
    const int* dst = ei + E;
    float* out = (float*)d_out;

    // ---- workspace carving (256B aligned) ----
    char* ws = (char*)d_ws;
    size_t off = 0;
    auto carve = [&](size_t bytes) -> char* {
        char* p = ws + off;
        off = (off + bytes + 255) & ~(size_t)255;
        return p;
    };
    float* normw = (float*)carve((size_t)E * 4);
    float* dinv  = (float*)carve((size_t)N * 4);   // deg -> dinv in place
    float* cinv  = (float*)carve((size_t)N * 4);   // cnt -> cnt_inv in place
    float* ssum  = (float*)carve((size_t)H * 4);
    float* svar  = (float*)carve((size_t)H * 4);
    unsigned short* Wb1 = (unsigned short*)carve((size_t)4 * FIN * H * 2);  // [4][H][FIN]
    unsigned short* Wb2 = (unsigned short*)carve((size_t)4 * H * H * 2);    // [4][H][H]
    unsigned short* Wb3 = (unsigned short*)carve((size_t)4 * H * Cc * 2);   // [4][Cc][H]
    unsigned short* hb  = (unsigned short*)carve((size_t)N * H * 2);
    float* bufA = (float*)carve((size_t)N * H * 4);
    float* bufB = (float*)carve((size_t)N * H * 4);
    float* bufC = (float*)carve((size_t)N * H * 4);

    auto cdiv = [](long a, long b) { return (int)((a + b - 1) / b); };

    // ---- edge norm precompute ----
    hipMemsetAsync(dinv, 0, (size_t)N * 4, stream);
    hipMemsetAsync(cinv, 0, (size_t)N * 4, stream);
    k_deg_cnt<<<cdiv(E, 256), 256, 0, stream>>>(w, dst, dinv, cinv, E);
    k_inv<<<cdiv(N, 256), 256, 0, stream>>>(dinv, cinv, N);
    k_norm_w<<<cdiv(E, 256), 256, 0, stream>>>(normw, dinv, w, src, dst, E);

    // ---- one-time bf16 conversions (weights transposed per slice) ----
    k_cvt_bf16_T<<<cdiv((long)4 * FIN * H, 256), 256, 0, stream>>>(W1, Wb1, FIN, H, (long)4 * FIN * H);
    k_cvt_bf16_T<<<cdiv((long)4 * H * H, 256), 256, 0, stream>>>(W2, Wb2, H, H, (long)4 * H * H);
    k_cvt_bf16_T<<<cdiv((long)4 * H * Cc, 256), 256, 0, stream>>>(W3, Wb3, H, Cc, (long)4 * H * Cc);
    k_cvt_bf16<<<cdiv((long)N * FIN, 256), 256, 0, stream>>>(x, hb, (long)N * FIN);

    auto gemm = [&](const unsigned short* A, const unsigned short* BT, float* Cp,
                    const float* bias, int M, int Nn, int Kd, int init) {
        dim3 grid(cdiv(M, TB_BM), cdiv(Nn, TB_BN));
        k_gemm_bf16<<<grid, 256, 0, stream>>>(A, BT, Cp, bias, M, Nn, Kd, init);
    };
    auto hop = [&](const float* hin, float* tmp, int d) {
        hipMemsetAsync(tmp, 0, (size_t)N * d * 4, stream);
        k_scatter<<<E, 256, 0, stream>>>(hin, tmp, normw, src, dst, E, d);
        k_scale_convert<<<cdiv((long)N * d / 4, 256), 256, 0, stream>>>(tmp, cinv, hb, N, d);
    };
    auto gnorm = [&](float* hpre, const float* gg, const float* bb, const float* aa,
                     float* houtF) {
        k_elu<<<cdiv((long)N * H, 256), 256, 0, stream>>>(hpre, (long)N * H);
        hipMemsetAsync(ssum, 0, (size_t)H * 4, stream);
        hipMemsetAsync(svar, 0, (size_t)H * 4, stream);
        dim3 grid(H / 256, cdiv(N, GN_ROWS));
        k_colsum<<<grid, 256, 0, stream>>>(hpre, ssum, N, H);
        k_colvar<<<grid, 256, 0, stream>>>(hpre, ssum, aa, svar, N, H);
        k_gnorm_apply<<<cdiv((long)N * H, 256), 256, 0, stream>>>(
            hpre, ssum, svar, gg, bb, aa, houtF, hb, N, H);
    };

    // ---- layer 1: TAGConv(128 -> 1024), hops in 128-dim space ----
    gemm(hb, Wb1 + (size_t)0 * FIN * H, bufC, b1, N, H, FIN, 1);
    hop(x, bufB, FIN);
    gemm(hb, Wb1 + (size_t)1 * FIN * H, bufC, b1, N, H, FIN, 0);
    hop(bufB, bufA, FIN);
    gemm(hb, Wb1 + (size_t)2 * FIN * H, bufC, b1, N, H, FIN, 0);
    hop(bufA, bufB, FIN);
    gemm(hb, Wb1 + (size_t)3 * FIN * H, bufC, b1, N, H, FIN, 0);
    gnorm(bufC, g1, be1, a1, bufA);       // bufA = layer2 input f32, hb = bf16

    // ---- layer 2: TAGConv(1024 -> 1024) ----
    gemm(hb, Wb2 + (size_t)0 * H * H, bufC, b2, N, H, H, 1);
    hop(bufA, bufB, H);
    gemm(hb, Wb2 + (size_t)1 * H * H, bufC, b2, N, H, H, 0);
    hop(bufB, bufA, H);
    gemm(hb, Wb2 + (size_t)2 * H * H, bufC, b2, N, H, H, 0);
    hop(bufA, bufB, H);
    gemm(hb, Wb2 + (size_t)3 * H * H, bufC, b2, N, H, H, 0);
    gnorm(bufC, g2, be2, a2, bufA);       // bufA = layer3 input f32, hb = bf16

    // ---- layer 3: TAGConv(1024 -> 16) -> d_out ----
    gemm(hb, Wb3 + (size_t)0 * H * Cc, out, b3, N, Cc, H, 1);
    hop(bufA, bufB, H);
    gemm(hb, Wb3 + (size_t)1 * H * Cc, out, b3, N, Cc, H, 0);
    hop(bufB, bufA, H);
    gemm(hb, Wb3 + (size_t)2 * H * Cc, out, b3, N, Cc, H, 0);
    hop(bufA, bufB, H);
    gemm(hb, Wb3 + (size_t)3 * H * Cc, out, b3, N, Cc, H, 0);
}
